// autoencoder_rnn2_15848429322983
// MI455X (gfx1250) — matrix operand order repaired and checked by reading, not compiled, not run on hardware
//
#include <hip/hip_runtime.h>
#include <hip/hip_bf16.h>

// ---------------------------------------------------------------------------
// MI455X (gfx1250) bf16-WMMA LSTM autoencoder.
//   - all weights converted to bf16 once per call (L2-resident, ~88MB < 192MB L2)
//   - every matmul is a GEMV done with v_wmma_f32_16x16x32_bf16 (f32 accum)
//   - activation vector staged in LDS once per block, B fed via ds_load_b128
//   - fwd/rev directions fused via blockIdx.y; time steps via per-step launches
// ---------------------------------------------------------------------------

typedef __bf16 v16bf __attribute__((ext_vector_type(16)));
typedef __bf16 v8bf  __attribute__((ext_vector_type(8)));
typedef float  v8f   __attribute__((ext_vector_type(8)));

#define T_SEQ 256
#define E_DIM 1024
#define H_DIM 2048
#define KMAX  4096   // largest GEMV K (cls: H+H) -> 8KB LDS staging buffer

// d_out flat offsets (floats)
#define O_LATF 0
#define O_LATR 640
#define O_RECF 1280
#define O_RECR 263424
#define O_EMBF 525568
#define O_EMBR 787712

__device__ __forceinline__ unsigned short f2bf(float f) {
    unsigned int u = __float_as_uint(f);
    u += 0x7FFFu + ((u >> 16) & 1u);           // round-to-nearest-even
    return (unsigned short)(u >> 16);
}
__device__ __forceinline__ float sigm(float x) { return 1.0f / (1.0f + expf(-x)); }

// ---------------------------------------------------------------------------
// f32 -> bf16 conversion with optional row-wise concatenation [A|B]
// ---------------------------------------------------------------------------
__global__ void convert_cat(const float* __restrict__ A, const float* __restrict__ B,
                            unsigned short* __restrict__ out, int M, int K1, int K2) {
    const size_t Kt = (size_t)K1 + K2;
    const size_t total = (size_t)M * Kt;
    for (size_t idx = (size_t)blockIdx.x * blockDim.x + threadIdx.x; idx < total;
         idx += (size_t)gridDim.x * blockDim.x) {
        size_t m = idx / Kt;
        size_t k = idx % Kt;
        float v = (k < (size_t)K1) ? A[m * K1 + k] : B[m * K2 + (k - K1)];
        out[idx] = f2bf(v);
    }
}

// ---------------------------------------------------------------------------
// Embedding gather: emb_f[t] = emb[x[t]], emb_r[t] = emb[x[T-1-t]]
// writes f32 to d_out and bf16 copies to workspace
// ---------------------------------------------------------------------------
__global__ void embed_kernel(const int* __restrict__ x, const float* __restrict__ emb,
                             float* __restrict__ outF, float* __restrict__ outR,
                             unsigned short* __restrict__ bfF, unsigned short* __restrict__ bfR) {
    const int t  = blockIdx.x;
    const int xf = x[t];
    const int xr = x[T_SEQ - 1 - t];
    for (int j = threadIdx.x; j < E_DIM; j += blockDim.x) {
        float vf = emb[xf * E_DIM + j];
        float vr = emb[xr * E_DIM + j];
        outF[t * E_DIM + j] = vf;  bfF[t * E_DIM + j] = f2bf(vf);
        outR[t * E_DIM + j] = vr;  bfR[t * E_DIM + j] = f2bf(vr);
    }
}

__global__ void init_enc(float* cF, float* cR, unsigned short* hFbf, unsigned short* hRbf) {
    int i = blockIdx.x * blockDim.x + threadIdx.x;
    if (i < H_DIM) { cF[i] = 0.f; cR[i] = 0.f; hFbf[i] = 0; hRbf[i] = 0; }
}

// ---------------------------------------------------------------------------
// GEMV via WMMA bf16: out[m] = sum_k W[m, k] * v[k] (+bias0 +bias1, opt tanh)
// v = concat(x[K1], h[K2]) staged in LDS once per block.
// A layout (16-bit A 16x32, ISA 7.12.2):
//   lane m (0-15):  a[0..7]=K0..7,  a[8..15]=K16..23
//   lane m+16:      a[0..7]=K8..15, a[8..15]=K24..31
// B 32x16: same k order as A (lanes 0-15 K=0..7,16..23; lanes 16-31 K=8..15,24..31), col=lane;
//   we broadcast v into all 16 columns so every column of D equals W@v.
// D (32-bit C/D 16x16): lane 0 holds M=0..7 in acc[0..7], lane 16 holds M=8..15.
// ---------------------------------------------------------------------------
__global__ __launch_bounds__(256)
void gemv_wmma_dual(const unsigned short* __restrict__ W0, const unsigned short* __restrict__ W1,
                    const unsigned short* __restrict__ x0, const unsigned short* __restrict__ x1,
                    const unsigned short* __restrict__ h0p, const unsigned short* __restrict__ h1p,
                    const float* b0a, const float* b0b,
                    const float* b1a, const float* b1b,
                    float* out0, float* out1,
                    unsigned short* bfo0, unsigned short* bfo1,
                    int K1, int K2, int act) {
    __shared__ __align__(16) unsigned short vlds[KMAX];

    const int dir = blockIdx.y;
    const unsigned short* W    = dir ? W1  : W0;
    const unsigned short* x    = dir ? x1  : x0;
    const unsigned short* hseg = dir ? h1p : h0p;
    const float* bias0 = dir ? b0b : b0a;
    const float* bias1 = dir ? b1b : b1a;
    float*          out = dir ? out1 : out0;
    unsigned short* bfo = dir ? bfo1 : bfo0;

    const int Ktot = K1 + K2;

    // ---- stage v = [x | h] into LDS (16B chunks; K1 is a multiple of 8) ----
    for (int k8 = threadIdx.x * 8; k8 < Ktot; k8 += blockDim.x * 8) {
        const unsigned short* src = (k8 < K1) ? (x + k8) : (hseg + (k8 - K1));
        *reinterpret_cast<v8bf*>(&vlds[k8]) = *reinterpret_cast<const v8bf*>(src);
    }
    __syncthreads();

    const int lane = threadIdx.x & 31;
    const int wave = threadIdx.x >> 5;
    const int row  = lane & 15;
    const int half = lane >> 4;
    const int rowBase = (blockIdx.x * 8 + wave) * 16;

    const unsigned short* wrow = W + (size_t)(rowBase + row) * Ktot;

    v8f acc = {};
    for (int k0 = 0; k0 < Ktot; k0 += 32) {
        if (k0 + 64 < Ktot) __builtin_prefetch(wrow + k0 + 64, 0, 0); // global_prefetch_b8

        // A tile: two 16B global loads per lane (weights stream from L2)
        v8bf alo = *reinterpret_cast<const v8bf*>(wrow + k0 + half * 8);
        v8bf ahi = *reinterpret_cast<const v8bf*>(wrow + k0 + 16 + half * 8);
        v16bf a = __builtin_shufflevector(alo, ahi, 0,1,2,3,4,5,6,7,8,9,10,11,12,13,14,15);

        // B: v[k] in the same k order as A (K 8h..8h+7, 16+8h..16+8h+7), from LDS (ds_load_b128)
        const int kb = k0 + half * 8;
        v8bf blo = *reinterpret_cast<const v8bf*>(&vlds[kb]);
        v8bf bhi = *reinterpret_cast<const v8bf*>(&vlds[kb + 16]);
        v16bf b = __builtin_shufflevector(blo, bhi, 0,1,2,3,4,5,6,7,8,9,10,11,12,13,14,15);

        acc = __builtin_amdgcn_wmma_f32_16x16x32_bf16(
            /*neg_a=*/false, a, /*neg_b=*/false, b,
            /*c_mod=*/(short)0, acc, /*reuse_a=*/false, /*reuse_b=*/false);
    }

    // All D columns are identical; lanes 0 and 16 own rows 0-7 / 8-15.
    if (row == 0) {
        const int mbase = rowBase + half * 8;
#pragma unroll
        for (int i = 0; i < 8; ++i) {
            float v = acc[i];
            if (bias0) v += bias0[mbase + i];
            if (bias1) v += bias1[mbase + i];
            if (act)   v  = tanhf(v);
            out[mbase + i] = v;
            if (bfo) bfo[mbase + i] = f2bf(v);
        }
    }
}

// ---------------------------------------------------------------------------
// Encoder gate nonlinearity: c,h update from g[4H]
// ---------------------------------------------------------------------------
__global__ void enc_gate_dual(const float* __restrict__ gF, const float* __restrict__ gR,
                              float* cF, float* cR,
                              unsigned short* hFbf, unsigned short* hRbf) {
    const int dir = blockIdx.y;
    const float* g = dir ? gR : gF;
    float* c = dir ? cR : cF;
    unsigned short* hb = dir ? hRbf : hFbf;
    const int m = blockIdx.x * blockDim.x + threadIdx.x;
    if (m >= H_DIM) return;
    float ig = g[m], fg = g[H_DIM + m], gg = g[2 * H_DIM + m], og = g[3 * H_DIM + m];
    float cn = sigm(fg) * c[m] + sigm(ig) * tanhf(gg);
    float hn = sigm(og) * tanhf(cn);
    c[m] = cn;
    hb[m] = f2bf(hn);
}

// ---------------------------------------------------------------------------
// Decoder gate: update c, write h into BOTH halves of v (prev==h after step 1),
// and scatter h into the reversed output sequence: rec[254 - step] = h
// ---------------------------------------------------------------------------
__global__ void dec_gate_dual(const float* __restrict__ gF, const float* __restrict__ gR,
                              float* cF, float* cR,
                              unsigned short* vF, unsigned short* vR,
                              float* recF, float* recR, int step) {
    const int dir = blockIdx.y;
    const float* g = dir ? gR : gF;
    float* c = dir ? cR : cF;
    unsigned short* v = dir ? vR : vF;
    float* rec = dir ? recR : recF;
    const int m = blockIdx.x * blockDim.x + threadIdx.x;
    if (m >= E_DIM) return;
    float ig = g[m], fg = g[E_DIM + m], gg = g[2 * E_DIM + m], og = g[3 * E_DIM + m];
    float cn = sigm(fg) * c[m] + sigm(ig) * tanhf(gg);
    float hn = sigm(og) * tanhf(cn);
    c[m] = cn;
    unsigned short hb = f2bf(hn);
    v[m] = hb;            // prev segment
    v[E_DIM + m] = hb;    // h segment
    rec[(254 - step) * E_DIM + m] = hn;
}

// ---------------------------------------------------------------------------
// Head: build lat_*_full, decoder initial states h0f/h0r (= c and v halves),
// and the boundary rows (start/end) of both reversed rec outputs.
// ---------------------------------------------------------------------------
__global__ void head_kernel(const int* Vg, const int* Jg,
                            const float* __restrict__ emb_Vg, const float* __restrict__ emb_Jg,
                            const float* __restrict__ emb,
                            const float* __restrict__ latf, const float* __restrict__ latr,
                            const float* __restrict__ mixv,
                            float* outLatF, float* outLatR,
                            float* recF, float* recR,
                            float* cF, float* cR,
                            unsigned short* vF, unsigned short* vR) {
    const int vgi = Vg[0], jgi = Jg[0];
    for (int i = threadIdx.x; i < E_DIM; i += blockDim.x) {
        float hf, hr;
        if (i < 64)       { hf = emb_Vg[vgi * 64 + i];        hr = hf; }
        else if (i < 576) { hf = latf[i - 64];                hr = latr[i - 64]; }
        else if (i < 640) { hf = emb_Jg[jgi * 64 + (i - 576)]; hr = hf; }
        else              { hf = mixv[i - 640];               hr = hf; }
        if (i < 640) { outLatF[i] = hf; outLatR[i] = hr; }
        cF[i] = hf;  cR[i] = hr;                 // c0 = h0c
        vF[E_DIM + i] = f2bf(hf);                // h0 segment
        vR[E_DIM + i] = f2bf(hr);
        float e0 = emb[0 * E_DIM + i];           // emb[X_IDX]
        float e1 = emb[1 * E_DIM + i];           // emb[B_IDX]
        vF[i] = f2bf(e0);                        // prev0 = start (fwd: emb[X])
        vR[i] = f2bf(e1);                        // prev0 = start (rev: emb[B])
        // rec = seq[::-1]: rec[255] = start_vec, rec[0] = end_vec
        recF[255 * E_DIM + i] = e0;  recF[i] = e1;
        recR[255 * E_DIM + i] = e1;  recR[i] = e0;
    }
}

// ---------------------------------------------------------------------------
extern "C" void kernel_launch(void* const* d_in, const int* in_sizes, int n_in,
                              void* d_out, int out_size, void* d_ws, size_t ws_size,
                              hipStream_t stream) {
    (void)in_sizes; (void)n_in; (void)out_size; (void)ws_size;

    const int*   x        = (const int*)d_in[0];
    const int*   Vg       = (const int*)d_in[1];
    const int*   Jg       = (const int*)d_in[2];
    const float* emb      = (const float*)d_in[3];
    const float* emb_Vg   = (const float*)d_in[4];
    const float* emb_Jg   = (const float*)d_in[5];
    const float* enc_Wih  = (const float*)d_in[6];
    const float* enc_Whh  = (const float*)d_in[7];
    const float* enc_bih  = (const float*)d_in[8];
    const float* enc_bhh  = (const float*)d_in[9];
    const float* cls_W    = (const float*)d_in[10];
    const float* cls_b    = (const float*)d_in[11];
    const float* latf_W   = (const float*)d_in[12];
    const float* latf_b   = (const float*)d_in[13];
    const float* latr_W   = (const float*)d_in[14];
    const float* latr_b   = (const float*)d_in[15];
    const float* mix_W    = (const float*)d_in[16];
    const float* mix_b    = (const float*)d_in[17];
    const float* recf_Wih = (const float*)d_in[18];
    const float* recf_Whh = (const float*)d_in[19];
    const float* recf_bih = (const float*)d_in[20];
    const float* recf_bhh = (const float*)d_in[21];
    const float* recr_Wih = (const float*)d_in[22];
    const float* recr_Whh = (const float*)d_in[23];
    const float* recr_bih = (const float*)d_in[24];
    const float* recr_bhh = (const float*)d_in[25];

    float* out = (float*)d_out;

    // ---- workspace carve (256B aligned); total ~90 MiB ----
    char* ws = (char*)d_ws;
    size_t off = 0;
    auto alloc = [&](size_t bytes) -> char* {
        char* p = ws + off;
        off = (off + bytes + 255) & ~(size_t)255;
        return p;
    };
    unsigned short* encW  = (unsigned short*)alloc((size_t)8192 * 3072 * 2);
    unsigned short* recfW = (unsigned short*)alloc((size_t)4096 * 2048 * 2);
    unsigned short* recrW = (unsigned short*)alloc((size_t)4096 * 2048 * 2);
    unsigned short* clsW  = (unsigned short*)alloc((size_t)512 * 4096 * 2);
    unsigned short* latfW = (unsigned short*)alloc((size_t)512 * 2048 * 2);
    unsigned short* latrW = (unsigned short*)alloc((size_t)512 * 2048 * 2);
    unsigned short* mixW  = (unsigned short*)alloc((size_t)384 * 512 * 2);
    unsigned short* embFbf = (unsigned short*)alloc((size_t)T_SEQ * E_DIM * 2);
    unsigned short* embRbf = (unsigned short*)alloc((size_t)T_SEQ * E_DIM * 2);
    unsigned short* hFbf = (unsigned short*)alloc(H_DIM * 2);
    unsigned short* hRbf = (unsigned short*)alloc(H_DIM * 2);
    float* cF = (float*)alloc(H_DIM * 4);
    float* cR = (float*)alloc(H_DIM * 4);
    float* gF = (float*)alloc(4 * H_DIM * 4);
    float* gR = (float*)alloc(4 * H_DIM * 4);
    float* common = (float*)alloc(512 * 4);
    unsigned short* commonBf = (unsigned short*)alloc(512 * 2);
    float* latf = (float*)alloc(512 * 4);
    float* latr = (float*)alloc(512 * 4);
    float* mixv = (float*)alloc(384 * 4);
    float* cdF = (float*)alloc(E_DIM * 4);
    float* cdR = (float*)alloc(E_DIM * 4);
    unsigned short* vF = (unsigned short*)alloc(2 * E_DIM * 2);
    unsigned short* vR = (unsigned short*)alloc(2 * E_DIM * 2);
    float* gdF = (float*)alloc(4 * E_DIM * 4);
    float* gdR = (float*)alloc(4 * E_DIM * 4);

    const dim3 b256(256);

    // ---- weight conversion (bf16, row-concatenated [Wih|Whh]) ----
    convert_cat<<<dim3(2048), b256, 0, stream>>>(enc_Wih, enc_Whh, encW, 8192, 1024, 2048);
    convert_cat<<<dim3(1024), b256, 0, stream>>>(recf_Wih, recf_Whh, recfW, 4096, 1024, 1024);
    convert_cat<<<dim3(1024), b256, 0, stream>>>(recr_Wih, recr_Whh, recrW, 4096, 1024, 1024);
    convert_cat<<<dim3(512),  b256, 0, stream>>>(cls_W,  nullptr, clsW,  512, 4096, 0);
    convert_cat<<<dim3(256),  b256, 0, stream>>>(latf_W, nullptr, latfW, 512, 2048, 0);
    convert_cat<<<dim3(256),  b256, 0, stream>>>(latr_W, nullptr, latrW, 512, 2048, 0);
    convert_cat<<<dim3(64),   b256, 0, stream>>>(mix_W,  nullptr, mixW,  384, 512, 0);

    // ---- embeddings (also writes emb_f / emb_r outputs) ----
    embed_kernel<<<dim3(T_SEQ), b256, 0, stream>>>(x, emb, out + O_EMBF, out + O_EMBR,
                                                   embFbf, embRbf);
    init_enc<<<dim3(H_DIM / 256), b256, 0, stream>>>(cF, cR, hFbf, hRbf);

    // ---- encoder: 256 steps, fwd+rev fused (grid.y = 2) ----
    for (int t = 0; t < T_SEQ; ++t) {
        gemv_wmma_dual<<<dim3(8192 / 128, 2), b256, 0, stream>>>(
            encW, encW,
            embFbf + (size_t)t * E_DIM, embRbf + (size_t)t * E_DIM,
            hFbf, hRbf,
            enc_bih, enc_bih, enc_bhh, enc_bhh,
            gF, gR, nullptr, nullptr,
            E_DIM, H_DIM, 0);
        enc_gate_dual<<<dim3(H_DIM / 256, 2), b256, 0, stream>>>(gF, gR, cF, cR, hFbf, hRbf);
    }

    // ---- heads ----
    // common = tanh([hf, hr] @ cls_W.T + cls_b)   (also produce bf16 for mix)
    gemv_wmma_dual<<<dim3(512 / 128, 1), b256, 0, stream>>>(
        clsW, clsW, hFbf, hFbf, hRbf, hRbf,
        cls_b, cls_b, nullptr, nullptr,
        common, common, commonBf, commonBf,
        H_DIM, H_DIM, 1);
    // lat_f = hf @ latf_W.T + latf_b ; lat_r = hr @ latr_W.T + latr_b (dual)
    gemv_wmma_dual<<<dim3(512 / 128, 2), b256, 0, stream>>>(
        latfW, latrW, hFbf, hRbf, nullptr, nullptr,
        latf_b, latr_b, nullptr, nullptr,
        latf, latr, nullptr, nullptr,
        H_DIM, 0, 0);
    // mix = common @ mix_W.T + mix_b
    gemv_wmma_dual<<<dim3(384 / 128, 1), b256, 0, stream>>>(
        mixW, mixW, commonBf, commonBf, nullptr, nullptr,
        mix_b, mix_b, nullptr, nullptr,
        mixv, mixv, nullptr, nullptr,
        512, 0, 0);

    head_kernel<<<dim3(1), b256, 0, stream>>>(Vg, Jg, emb_Vg, emb_Jg, emb,
                                              latf, latr, mixv,
                                              out + O_LATF, out + O_LATR,
                                              out + O_RECF, out + O_RECR,
                                              cdF, cdR, vF, vR);

    // ---- decoders: 254 steps, fwd+rev fused ----
    for (int j = 0; j < T_SEQ - 2; ++j) {
        gemv_wmma_dual<<<dim3(4096 / 128, 2), b256, 0, stream>>>(
            recfW, recrW, vF, vR, nullptr, nullptr,
            recf_bih, recr_bih, recf_bhh, recr_bhh,
            gdF, gdR, nullptr, nullptr,
            2 * E_DIM, 0, 0);
        dec_gate_dual<<<dim3(E_DIM / 256, 2), b256, 0, stream>>>(
            gdF, gdR, cdF, cdR, vF, vR,
            out + O_RECF, out + O_RECR, j);
    }
}
